// GCN_DeepSet_AntiSym_Invariant_34565896798211
// MI455X (gfx1250) — matrix operand order repaired and checked by reading, not compiled, not run on hardware
//
#include <hip/hip_runtime.h>

// ---------------------------------------------------------------------------
// Problem constants (match reference)
// ---------------------------------------------------------------------------
#define Bb      4096
#define Nn      10
#define DINc    32
#define Dm      128
#define PHIDIM  256
#define RHODIM  128
#define Lc      3
#define BNc     (Bb * Nn)          // 40960
#define EPSc    1e-5f
#define K3c     (3 * Dm)           // 384: stacked RGCN K
#define LDSPAD  8                  // bf16 row pad -> 4-bank stride, conflict-free

typedef __attribute__((ext_vector_type(16))) __bf16 v16bf;
typedef __attribute__((ext_vector_type(8)))  __bf16 v8bf;
typedef __attribute__((ext_vector_type(8)))  float  v8f;

// ---------------------------------------------------------------------------
// CDNA5 async global->LDS copy (ASYNCcnt-tracked, no VGPR round trip).
// ISA 15.18.3: GLOBAL_LOAD_ASYNC_TO_LDS_B128, VDST = LDS byte offset VGPR.
// ---------------------------------------------------------------------------
__device__ __forceinline__ void async_copy_b128(const void* gsrc, unsigned lds_off) {
#if defined(__gfx1250__)
    unsigned long long ga = (unsigned long long)(size_t)gsrc;
    asm volatile("global_load_async_to_lds_b128 %0, %1, off"
                 :: "v"(lds_off), "v"(ga) : "memory");
#else
    (void)gsrc; (void)lds_off;
#endif
}

__device__ __forceinline__ void wait_async_zero() {
#if defined(__gfx1250__)
    asm volatile("s_wait_asynccnt 0" ::: "memory");
#endif
}

// Stage Wt[Nout, K] bf16 into LDS with +LDSPAD/row padding. 16B per lane per op.
template <int K>
__device__ __forceinline__ void stage_weights(const __bf16* __restrict__ Wt,
                                              __bf16* smem, int Nout) {
    constexpr int ldK = K + LDSPAD;
    constexpr int CPR = K / 8;                 // 16B chunks per row
    const int chunks = Nout * CPR;
    for (int c = threadIdx.x; c < chunks; c += 256) {
        const int n = c / CPR;
        const int w = c - n * CPR;
        const unsigned dst = (unsigned)(size_t)(smem + n * ldK + w * 8);
        async_copy_b128(Wt + (size_t)n * K + w * 8, dst);
    }
    wait_async_zero();
    __syncthreads();
}

// B fragment from LDS: two 8-K runs at p and p+16 (two aligned 16B ds reads).
__device__ __forceinline__ v16bf load_b_frag(const __bf16* p) {
    const v8bf lo = *(const v8bf*)(p);
    const v8bf hi = *(const v8bf*)(p + 16);
    v16bf r;
#pragma unroll
    for (int i = 0; i < 8; ++i) { r[i] = lo[i]; r[i + 8] = hi[i]; }
    return r;
}

// A fragment: 16 rows x 32 K of f32 -> bf16 regs.
// ISA 7.12.2 layout: lane l -> row l%16; elem i -> K = kb + (i<8?0:16) + 8*(l/16) + i%8
// == two contiguous 8-float runs.
__device__ __forceinline__ v16bf load_a_frag(const float* arow, int kb) {
    const float4 f0 = *(const float4*)(arow + kb);
    const float4 f1 = *(const float4*)(arow + kb + 4);
    const float4 f2 = *(const float4*)(arow + kb + 16);
    const float4 f3 = *(const float4*)(arow + kb + 20);
    v16bf a;
    a[0]  = (__bf16)f0.x; a[1]  = (__bf16)f0.y; a[2]  = (__bf16)f0.z; a[3]  = (__bf16)f0.w;
    a[4]  = (__bf16)f1.x; a[5]  = (__bf16)f1.y; a[6]  = (__bf16)f1.z; a[7]  = (__bf16)f1.w;
    a[8]  = (__bf16)f2.x; a[9]  = (__bf16)f2.y; a[10] = (__bf16)f2.z; a[11] = (__bf16)f2.w;
    a[12] = (__bf16)f3.x; a[13] = (__bf16)f3.y; a[14] = (__bf16)f3.z; a[15] = (__bf16)f3.w;
    return a;
}

// ---------------------------------------------------------------------------
// Weight convert + transpose with stacking support:
//   Wt[n*ldt + koff + k] = (bf16) W[k*Nout + n]
// ---------------------------------------------------------------------------
__global__ void wcvt_kernel(const float* __restrict__ W, __bf16* __restrict__ Wt,
                            int K, int Nout, int ldt, int koff) {
    int idx = blockIdx.x * blockDim.x + threadIdx.x;
    if (idx >= K * Nout) return;
    int k = idx / Nout;
    int n = idx - k * Nout;
    Wt[(size_t)n * ldt + koff + k] = (__bf16)W[idx];
}

// ---------------------------------------------------------------------------
// Row-parallel GEMM: Out[M,Nout] = act(Xin[M,K] @ W + bias).
// 8 waves/block, 16 rows/wave; A preloaded to regs (wave-private),
// B staged block-wide in LDS via async copies (block-shared).
// ---------------------------------------------------------------------------
template <int K, bool RELU>
__global__ __launch_bounds__(256)
void gemm_bf16_wmma(const float* __restrict__ Xin,
                    const __bf16* __restrict__ Wt,
                    const float* __restrict__ bias,
                    float* __restrict__ Out,
                    int Nout) {
    constexpr int KT  = K / 32;
    constexpr int ldK = K + LDSPAD;
    extern __shared__ __align__(16) unsigned char smem_raw[];
    __bf16* smem = (__bf16*)smem_raw;

    stage_weights<K>(Wt, smem, Nout);

    const int lane = threadIdx.x & 31;
    const int wave = threadIdx.x >> 5;
    const int r0   = blockIdx.x * 128 + wave * 16;
    const int m    = lane & 15;
    const int half = lane >> 4;

    v16bf afrag[KT];
    const float* arow = Xin + (size_t)(r0 + m) * K;
#pragma unroll
    for (int kt = 0; kt < KT; ++kt)
        afrag[kt] = load_a_frag(arow, kt * 32 + half * 8);

    const int ntiles = Nout >> 4;
    for (int nt = 0; nt < ntiles; ++nt) {
        const int n = nt * 16 + (lane & 15);
        const float binit = bias[n];
        v8f acc;
#pragma unroll
        for (int g = 0; g < 8; ++g) acc[g] = binit;

        const __bf16* brow = smem + (size_t)n * ldK;
#pragma unroll
        for (int kt = 0; kt < KT; ++kt) {
            v16bf bf = load_b_frag(brow + kt * 32 + half * 8);
            acc = __builtin_amdgcn_wmma_f32_16x16x32_bf16(
                false, afrag[kt], false, bf, (short)0, acc, false, false);
        }
#pragma unroll
        for (int g = 0; g < 8; ++g) {
            float v = acc[g];
            if (RELU) v = fmaxf(v, 0.0f);
            Out[(size_t)(r0 + half * 8 + g) * Nout + n] = v;
        }
    }
}

// ---------------------------------------------------------------------------
// Fused RGCN GEMM: H2 = mean0@W0 + mean1@W1 + H@Wroot + b
// Stacked as one K=384 GEMM over Wt3[Nout=128, 384] = [Wt0 | Wt1 | Wroot^T].
// Removes two full read-modify-write passes over the 21 MB H2 buffer.
// ---------------------------------------------------------------------------
__global__ __launch_bounds__(256)
void gemm3_bf16_wmma(const float* __restrict__ X0,
                     const float* __restrict__ X1,
                     const float* __restrict__ X2,
                     const __bf16* __restrict__ Wt3,
                     const float* __restrict__ bias,
                     float* __restrict__ Out) {
    constexpr int Nout = Dm;
    constexpr int ldK  = K3c + LDSPAD;
    extern __shared__ __align__(16) unsigned char smem_raw[];
    __bf16* smem = (__bf16*)smem_raw;

    stage_weights<K3c>(Wt3, smem, Nout);

    const int lane = threadIdx.x & 31;
    const int wave = threadIdx.x >> 5;
    const int r0   = blockIdx.x * 128 + wave * 16;
    const int m    = lane & 15;
    const int half = lane >> 4;

    v16bf afrag[12];
    {
        const float* srcs[3] = {X0, X1, X2};
#pragma unroll
        for (int s = 0; s < 3; ++s) {
            const float* arow = srcs[s] + (size_t)(r0 + m) * Dm;
#pragma unroll
            for (int kt = 0; kt < 4; ++kt)
                afrag[s * 4 + kt] = load_a_frag(arow, kt * 32 + half * 8);
        }
    }

    for (int nt = 0; nt < (Nout >> 4); ++nt) {
        const int n = nt * 16 + (lane & 15);
        const float binit = bias[n];
        v8f acc;
#pragma unroll
        for (int g = 0; g < 8; ++g) acc[g] = binit;

        const __bf16* brow = smem + (size_t)n * ldK;
#pragma unroll
        for (int kt = 0; kt < 12; ++kt) {
            v16bf bf = load_b_frag(brow + kt * 32 + half * 8);
            acc = __builtin_amdgcn_wmma_f32_16x16x32_bf16(
                false, afrag[kt], false, bf, (short)0, acc, false, false);
        }
#pragma unroll
        for (int g = 0; g < 8; ++g)
            Out[(size_t)(r0 + half * 8 + g) * Nout + n] = acc[g];
    }
}

// ---------------------------------------------------------------------------
// y += x elementwise (residual adds)
// ---------------------------------------------------------------------------
__global__ void axpy_add_kernel(float* __restrict__ y, const float* __restrict__ x, int n) {
    int i = blockIdx.x * blockDim.x + threadIdx.x;
    if (i < n) y[i] += x[i];
}

// ---------------------------------------------------------------------------
// Per-relation means. Block = (b, j) target node; thread = channel d.
// ---------------------------------------------------------------------------
__global__ __launch_bounds__(Dm)
void relmean_kernel(const float* __restrict__ A, const float* __restrict__ H,
                    float* __restrict__ M0, float* __restrict__ M1) {
    const int bj = blockIdx.x;
    const int b  = bj / Nn;
    const int j  = bj - b * Nn;
    const int d  = threadIdx.x;
    float s0 = 0.f, s1 = 0.f;
    int c1 = 0;
#pragma unroll
    for (int i = 0; i < Nn; ++i) {
        const float a = A[((size_t)b * Nn + i) * Nn + j];
        const float h = H[((size_t)b * Nn + i) * Dm + d];
        if (a > 0.f) { s1 += h; ++c1; } else { s0 += h; }
    }
    const float c1f = (float)c1;
    const float c0f = (float)(Nn - c1);
    M0[(size_t)bj * Dm + d] = s0 / fmaxf(c0f, 1.f);
    M1[(size_t)bj * Dm + d] = s1 / fmaxf(c1f, 1.f);
}

// ---------------------------------------------------------------------------
// agg[b,i,:] = relu(layernorm(sum_j |A[b,i,j]| * H2[b,j,:]))
// ---------------------------------------------------------------------------
__global__ __launch_bounds__(Dm)
void agg_ln_kernel(const float* __restrict__ A, const float* __restrict__ H2,
                   const float* __restrict__ lng, const float* __restrict__ lnb,
                   float* __restrict__ AGG) {
    const int bi = blockIdx.x;
    const int b  = bi / Nn;
    const int i  = bi - b * Nn;
    const int d  = threadIdx.x;
    float acc = 0.f;
#pragma unroll
    for (int j = 0; j < Nn; ++j) {
        const float w = fabsf(A[((size_t)b * Nn + i) * Nn + j]);
        acc += w * H2[((size_t)b * Nn + j) * Dm + d];
    }
    __shared__ float sm[Dm];
    sm[d] = acc;
    __syncthreads();
    for (int s = Dm / 2; s > 0; s >>= 1) {
        if (d < s) sm[d] += sm[d + s];
        __syncthreads();
    }
    const float mu = sm[0] * (1.f / Dm);
    __syncthreads();
    const float df = acc - mu;
    sm[d] = df * df;
    __syncthreads();
    for (int s = Dm / 2; s > 0; s >>= 1) {
        if (d < s) sm[d] += sm[d + s];
        __syncthreads();
    }
    const float var = sm[0] * (1.f / Dm);
    const float v = df * rsqrtf(var + EPSc) * lng[d] + lnb[d];
    AGG[(size_t)bi * Dm + d] = fmaxf(v, 0.f);
}

// ---------------------------------------------------------------------------
// DeepSet masked pooling
// ---------------------------------------------------------------------------
__global__ __launch_bounds__(PHIDIM)
void deepset_kernel(const float* __restrict__ PHI, const int* __restrict__ hm,
                    float* __restrict__ HS, float* __restrict__ AWS) {
    const int b = blockIdx.x;
    const int c = threadIdx.x;
    float s = 0.f, t = 0.f;
#pragma unroll
    for (int n = 0; n < Nn; ++n) {
        const float m = (float)hm[b * Nn + n];
        const float p = PHI[((size_t)b * Nn + n) * PHIDIM + c];
        s += p * m;
        t += p * (1.f - m);
    }
    HS [(size_t)b * PHIDIM + c] = s;
    AWS[(size_t)b * PHIDIM + c] = t;
}

// ---------------------------------------------------------------------------
// Final: out[b] = 0.5 + 0.5*tanh( (t1[b]-t2[b]) . r2w )   (r2b cancels)
// ---------------------------------------------------------------------------
__global__ __launch_bounds__(RHODIM)
void final_kernel(const float* __restrict__ T1, const float* __restrict__ T2,
                  const float* __restrict__ r2w, float* __restrict__ out) {
    const int b = blockIdx.x;
    const int d = threadIdx.x;
    float v = (T1[(size_t)b * RHODIM + d] - T2[(size_t)b * RHODIM + d]) * r2w[d];
    __shared__ float sm[RHODIM];
    sm[d] = v;
    __syncthreads();
    for (int s = RHODIM / 2; s > 0; s >>= 1) {
        if (d < s) sm[d] += sm[d + s];
        __syncthreads();
    }
    if (d == 0) out[b] = 0.5f + 0.5f * tanhf(sm[0]);
}

// ---------------------------------------------------------------------------
// Host orchestration
// ---------------------------------------------------------------------------
extern "C" void kernel_launch(void* const* d_in, const int* in_sizes, int n_in,
                              void* d_out, int out_size, void* d_ws, size_t ws_size,
                              hipStream_t stream) {
    (void)in_sizes; (void)n_in; (void)out_size; (void)ws_size;

    const float* A    = (const float*)d_in[0];
    const float* X    = (const float*)d_in[1];
    const int*   hm   = (const int*)  d_in[2];
    const float* We1  = (const float*)d_in[3];
    const float* be1  = (const float*)d_in[4];
    const float* We2  = (const float*)d_in[5];
    const float* be2  = (const float*)d_in[6];
    const float* rgw  = (const float*)d_in[7];   // [L,2,D,D]
    const float* rgr  = (const float*)d_in[8];   // [L,D,D]
    const float* rgb  = (const float*)d_in[9];   // [L,D]
    const float* l1w  = (const float*)d_in[10];
    const float* l1b  = (const float*)d_in[11];
    const float* l2w  = (const float*)d_in[12];
    const float* l2b  = (const float*)d_in[13];
    const float* lng  = (const float*)d_in[14];
    const float* lnb  = (const float*)d_in[15];
    const float* p1w  = (const float*)d_in[16];
    const float* p1b  = (const float*)d_in[17];
    const float* p2w  = (const float*)d_in[18];
    const float* p2b  = (const float*)d_in[19];
    const float* r1w  = (const float*)d_in[20];
    const float* r1b  = (const float*)d_in[21];
    const float* r2w  = (const float*)d_in[22];
    float*       out  = (float*)d_out;

    unsigned char* base = (unsigned char*)d_ws;
    size_t off = 0;
    auto alloc = [&](size_t bytes) -> void* {
        void* p = base + off;
        off += (bytes + 255) & ~(size_t)255;
        return p;
    };

    const size_t DD = (size_t)Dm * Dm;
    __bf16* we1t = (__bf16*)alloc((size_t)DINc * Dm * 2);
    __bf16* we2t = (__bf16*)alloc(DD * 2);
    __bf16* rg3t[Lc];                    // stacked [Dm, 3*Dm]
    __bf16* l1t[Lc]; __bf16* l2t[Lc];
    for (int l = 0; l < Lc; ++l) {
        rg3t[l] = (__bf16*)alloc((size_t)Dm * K3c * 2);
        l1t[l]  = (__bf16*)alloc(DD * 2);
        l2t[l]  = (__bf16*)alloc(DD * 2);
    }
    __bf16* p1t = (__bf16*)alloc((size_t)Dm * PHIDIM * 2);
    __bf16* p2t = (__bf16*)alloc((size_t)PHIDIM * PHIDIM * 2);
    __bf16* r1t = (__bf16*)alloc((size_t)PHIDIM * RHODIM * 2);

    float* H   = (float*)alloc((size_t)BNc * Dm * 4);
    float* AGG = (float*)alloc((size_t)BNc * Dm * 4);
    float* M0  = (float*)alloc((size_t)BNc * Dm * 4);
    float* M1  = (float*)alloc((size_t)BNc * Dm * 4);
    float* H2  = (float*)alloc((size_t)BNc * Dm * 4);
    float* PHI = (float*)alloc((size_t)BNc * PHIDIM * 4);
    float* HS  = (float*)alloc((size_t)Bb * PHIDIM * 4);
    float* AWS = (float*)alloc((size_t)Bb * PHIDIM * 4);
    float* T1  = (float*)alloc((size_t)Bb * RHODIM * 4);
    float* T2  = (float*)alloc((size_t)Bb * RHODIM * 4);

    auto cvt = [&](const float* W, __bf16* Wt, int K, int Nout, int ldt, int koff) {
        const int n = K * Nout;
        wcvt_kernel<<<(n + 255) / 256, 256, 0, stream>>>(W, Wt, K, Nout, ldt, koff);
    };

    // --- weight conversion (transposed bf16 copies; RGCN stacked to K=384) ---
    cvt(We1, we1t, DINc, Dm, DINc, 0);
    cvt(We2, we2t, Dm, Dm, Dm, 0);
    for (int l = 0; l < Lc; ++l) {
        cvt(rgw + ((size_t)l * 2 + 0) * DD, rg3t[l], Dm, Dm, K3c, 0);
        cvt(rgw + ((size_t)l * 2 + 1) * DD, rg3t[l], Dm, Dm, K3c, Dm);
        cvt(rgr + (size_t)l * DD,           rg3t[l], Dm, Dm, K3c, 2 * Dm);
        cvt(l1w + (size_t)l * DD, l1t[l], Dm, Dm, Dm, 0);
        cvt(l2w + (size_t)l * DD, l2t[l], Dm, Dm, Dm, 0);
    }
    cvt(p1w, p1t, Dm, PHIDIM, Dm, 0);
    cvt(p2w, p2t, PHIDIM, PHIDIM, PHIDIM, 0);
    cvt(r1w, r1t, PHIDIM, RHODIM, PHIDIM, 0);

    const dim3 blk(256);
    const int gBN = BNc / 128;   // 320 blocks
    const int gB  = Bb  / 128;   // 32 blocks
    auto shb = [](int K, int Nout) -> size_t { return (size_t)(K + LDSPAD) * Nout * 2; };

    // --- embedder ---
    gemm_bf16_wmma<DINc, true ><<<gBN, blk, shb(DINc, Dm), stream>>>(X,  we1t, be1, H2, Dm);
    gemm_bf16_wmma<Dm,   false><<<gBN, blk, shb(Dm, Dm),   stream>>>(H2, we2t, be2, H,  Dm);

    // --- message-passing layers ---
    for (int l = 0; l < Lc; ++l) {
        if (l > 0)
            axpy_add_kernel<<<(BNc * Dm + 255) / 256, 256, 0, stream>>>(H, AGG, BNc * Dm);
        relmean_kernel<<<BNc, Dm, 0, stream>>>(A, H, M0, M1);
        gemm3_bf16_wmma<<<gBN, blk, shb(K3c, Dm), stream>>>(M0, M1, H, rg3t[l],
                                                            rgb + (size_t)l * Dm, H2);
        agg_ln_kernel<<<BNc, Dm, 0, stream>>>(A, H2, lng, lnb, AGG);
        gemm_bf16_wmma<Dm, true ><<<gBN, blk, shb(Dm, Dm), stream>>>(AGG, l1t[l], l1b + (size_t)l * Dm, H2,  Dm);
        gemm_bf16_wmma<Dm, false><<<gBN, blk, shb(Dm, Dm), stream>>>(H2,  l2t[l], l2b + (size_t)l * Dm, AGG, Dm);
    }
    axpy_add_kernel<<<(BNc * Dm + 255) / 256, 256, 0, stream>>>(H, AGG, BNc * Dm);

    // --- DeepSet phi ---
    gemm_bf16_wmma<Dm,     true><<<gBN, blk, shb(Dm, PHIDIM),     stream>>>(H,   p1t, p1b, PHI, PHIDIM);
    gemm_bf16_wmma<PHIDIM, true><<<gBN, blk, shb(PHIDIM, PHIDIM), stream>>>(PHI, p2t, p2b, PHI, PHIDIM);
    deepset_kernel<<<Bb, PHIDIM, 0, stream>>>(PHI, hm, HS, AWS);

    // --- rho head ---
    gemm_bf16_wmma<PHIDIM, true><<<gB, blk, shb(PHIDIM, RHODIM), stream>>>(HS,  r1t, r1b, T1, RHODIM);
    gemm_bf16_wmma<PHIDIM, true><<<gB, blk, shb(PHIDIM, RHODIM), stream>>>(AWS, r1t, r1b, T2, RHODIM);
    final_kernel<<<Bb, RHODIM, 0, stream>>>(T1, T2, r2w, out);
}
